// DynamicRNN_27230092657493
// MI455X (gfx1250) — matrix operand order repaired and checked by reading, not compiled, not run on hardware
//
#include <hip/hip_runtime.h>

// ---------------- problem constants ----------------
#define T_STEPS 1024
#define BATCH   64
#define INF     256
#define HID     512
#define KTOT    768          // INF + HID (fused input+recurrent GEMM)
#define OUTF    5
#define NWG     32           // compute workgroups (each owns 16 hidden cols)
#define NWG_TOTAL 33         // + 1 FC workgroup
#define HS      16           // hidden slice per WG
#define ROWSW   64           // 4 gates * HS weight rows per WG
#define PITCH   776          // W LDS row pitch (bf16): 1552B -> 16B bank skew
#define XPITCH  264          // x panel pitch (bf16): 528B  -> 16B bank skew
#define HPITCH  520          // h panel pitch (bf16): 1040B -> 16B bank skew (TDM pad)
#define THREADS 256
#define OUT_OFF_H (T_STEPS*BATCH*OUTF)            // 327680
#define OUT_OFF_C (OUT_OFF_H + BATCH*HID)         // 360448

// LDS byte offsets
#define LDS_WB   0
#define LDS_XB   (ROWSW * PITCH * 2)              // 99328
#define LDS_HB   (LDS_XB + BATCH * XPITCH * 2)    // 133120
#define LDS_GATE (LDS_HB + BATCH * HPITCH * 2)    // 199680
#define LDS_TOTAL (LDS_GATE + BATCH * 4 * HS * 4) // 216064

typedef __bf16 bf16_t;
typedef __attribute__((ext_vector_type(8)))  float        v8f;
typedef __attribute__((ext_vector_type(16))) bf16_t       v16bf;
typedef __attribute__((ext_vector_type(8)))  bf16_t       v8bf;
typedef __attribute__((ext_vector_type(4)))  unsigned int v4u;
typedef __attribute__((ext_vector_type(4)))  int          v4i;
typedef __attribute__((ext_vector_type(8)))  int          v8i;

union BFrag { v16bf v; v8bf h[2]; };

__device__ __forceinline__ float sigmoidf_fast(float x) {
    return 1.0f / (1.0f + __expf(-x));
}

// ---------------------------------------------------------------------------
// Tensor Data Mover: 2D tile load Global -> LDS with per-row LDS padding.
//   tile/tensor: tensor_d0 x tensor_d1 elements (2-byte), stride0 elems,
//   pad_code7=256 DWORDs interval etc. (D# layout per CDNA5 ISA ch.8)
// ---------------------------------------------------------------------------
__device__ __forceinline__ void tdm_load_2d_bf16(unsigned lds_addr,
                                                 const void* gptr,
                                                 unsigned tensor_d0,
                                                 unsigned tensor_d1,
                                                 unsigned stride0,
                                                 unsigned pad_interval_code,
                                                 unsigned pad_amount_code) {
    unsigned long long ga = (unsigned long long)gptr;
    v4u g0;
    g0[0] = 1u;                                        // count=1 (valid), user mode
    g0[1] = lds_addr;                                  // LDS byte address
    g0[2] = (unsigned)(ga & 0xffffffffu);              // global_addr[31:0]
    g0[3] = (unsigned)((ga >> 32) & 0x1ffffffu)        // global_addr[56:32]
          | (2u << 30);                                // type=2 ("image")
    v8i g1;
    g1[0] = (int)((1u << 16)                           // data_size=2 bytes
                | (1u << 20)                           // pad_enable
                | (pad_interval_code << 22)
                | (pad_amount_code << 25));
    g1[1] = (int)((tensor_d0 & 0xffffu) << 16);        // tensor_dim0[15:0]
    g1[2] = (int)((tensor_d0 >> 16)                    // tensor_dim0[31:16]
                | ((tensor_d1 & 0xffffu) << 16));      // tensor_dim1[15:0]
    g1[3] = (int)((tensor_d1 >> 16)
                | ((tensor_d0 & 0xffffu) << 16));      // tile_dim0 = tensor_d0
    g1[4] = (int)(tensor_d1 & 0xffffu);                // tile_dim1 = tensor_d1, tile_dim2=0
    g1[5] = (int)stride0;                              // tensor_dim0_stride[31:0]
    g1[6] = 0;                                         // stride0[47:32], dim1_stride lo
    g1[7] = 0;
    v4i z4 = {0, 0, 0, 0};                             // 2-D tensor: groups 2/3 unused
#if defined(__clang_major__) && (__clang_major__ >= 23)
    v8i z8 = {0, 0, 0, 0, 0, 0, 0, 0};
    __builtin_amdgcn_tensor_load_to_lds(g0, g1, z4, z4, z8, 0);
#else
    __builtin_amdgcn_tensor_load_to_lds(g0, g1, z4, z4, 0);
#endif
}

// Sense-free monotonic-phase grid barrier across NWG_TOTAL blocks.
__device__ __forceinline__ void grid_barrier(unsigned* cnt, unsigned* phase,
                                             unsigned target) {
    __syncthreads();
    if (threadIdx.x == 0) {
        __threadfence();
        unsigned prev = __hip_atomic_fetch_add(cnt, 1u, __ATOMIC_ACQ_REL,
                                               __HIP_MEMORY_SCOPE_AGENT);
        if (prev == NWG_TOTAL - 1) {
            __hip_atomic_store(cnt, 0u, __ATOMIC_RELAXED, __HIP_MEMORY_SCOPE_AGENT);
            __hip_atomic_fetch_add(phase, 1u, __ATOMIC_RELEASE,
                                   __HIP_MEMORY_SCOPE_AGENT);
        } else {
            while (__hip_atomic_load(phase, __ATOMIC_ACQUIRE,
                                     __HIP_MEMORY_SCOPE_AGENT) < target) {
                __builtin_amdgcn_s_sleep(2);
            }
        }
    }
    __syncthreads();
}

// Pack W_ih(2048x256) and W_hh(2048x512) f32 -> fused bf16 W[2048][768];
// zero h0 buffer and barrier state. Runs every call (deterministic).
__global__ void lstm_prep(const float* __restrict__ Wih,
                          const float* __restrict__ Whh,
                          bf16_t* __restrict__ Wcomb,
                          bf16_t* __restrict__ h0,
                          unsigned* __restrict__ bar) {
    int idx = blockIdx.x * blockDim.x + threadIdx.x;
    const int NW = 4 * HID * KTOT;            // 2048*768
    if (idx < NW) {
        int r = idx / KTOT, c = idx % KTOT;
        float v = (c < INF) ? Wih[r * INF + c] : Whh[r * HID + (c - INF)];
        Wcomb[idx] = (bf16_t)v;
    }
    if (idx < BATCH * HID) h0[idx] = (bf16_t)0.0f;
    if (idx < 2) bar[idx] = 0u;
}

__launch_bounds__(THREADS)
__global__ void lstm_persistent(const float* __restrict__ x,     // [T,B,I]
                                const float* __restrict__ bih,   // [4H]
                                const float* __restrict__ bhh,   // [4H]
                                const float* __restrict__ fcw,   // [O,H]
                                const float* __restrict__ fcb,   // [O]
                                const bf16_t* __restrict__ Wcomb,// [4H,KTOT]
                                bf16_t* __restrict__ hbuf0,      // [B,H]
                                bf16_t* __restrict__ hbuf1,      // [B,H]
                                unsigned* __restrict__ bar,
                                float* __restrict__ out) {
    extern __shared__ char smem[];
    bf16_t* Wb     = (bf16_t*)(smem + LDS_WB);    // 64 x PITCH  (weights, resident)
    bf16_t* Xb     = (bf16_t*)(smem + LDS_XB);    // 64 x XPITCH (x_t panel)
    bf16_t* Hb     = (bf16_t*)(smem + LDS_HB);    // 64 x HPITCH (h_t panel, TDM)
    float*  gatesL = (float*)(smem + LDS_GATE);   // [64][4][16] f32

    const unsigned tid = threadIdx.x;
    const unsigned wg  = blockIdx.x;
    unsigned* cnt   = bar;
    unsigned* phase = bar + 1;
    unsigned target = 0;

    // ------------------------------------------------------------------
    // FC workgroup: computes out[t-1] = h_t @ fc_w^T + fc_b one step
    // behind the recurrence (fully overlapped with the next GEMM step).
    // ------------------------------------------------------------------
    if (wg == NWG) {
        for (int t = 0; t < T_STEPS; ++t) {
            const bf16_t* hc = (t & 1) ? hbuf1 : hbuf0;   // holds h_t
            if (t > 0 && tid < BATCH * OUTF) {
                int b = tid / OUTF, o = tid % OUTF;
                const bf16_t* hr = hc + b * HID;
                const float*  wr = fcw + o * HID;
                float s = fcb[o];
                for (int j = 0; j < HID; ++j) s += (float)hr[j] * wr[j];
                out[((size_t)(t - 1) * BATCH + b) * OUTF + o] = s;
            }
            grid_barrier(cnt, phase, ++target);
        }
        // T even -> h_T lives in hbuf0
        if (tid < BATCH * OUTF) {
            int b = tid / OUTF, o = tid % OUTF;
            const bf16_t* hr = hbuf0 + b * HID;
            const float*  wr = fcw + o * HID;
            float s = fcb[o];
            for (int j = 0; j < HID; ++j) s += (float)hr[j] * wr[j];
            out[((size_t)(T_STEPS - 1) * BATCH + b) * OUTF + o] = s;
        }
        return;
    }

    // ------------------------------------------------------------------
    // Compute workgroup wg owns hidden cols [jbase, jbase+16).
    // Weight rows (i,f,g,o gates x 16 cols) = 64 rows x 768 K in LDS forever.
    // ------------------------------------------------------------------
    const int jbase = wg * HS;
    for (int i = tid; i < ROWSW * (KTOT / 8); i += THREADS) {   // 6144 uint4
        int lr = i / (KTOT / 8), ch = i % (KTOT / 8);
        int g = lr >> 4, jj = lr & 15;
        int rglob = g * HID + jbase + jj;
        uint4 v = *(const uint4*)(Wcomb + (size_t)rglob * KTOT + ch * 8);
        *(uint4*)(Wb + lr * PITCH + ch * 8) = v;
    }

    const int wv   = tid >> 5;        // wave id 0..7 (wave32)
    const int lane = tid & 31;
    const int hsel = lane >> 4;       // lane half-select for WMMA layouts
    const int lm   = lane & 15;
    const int mtile = wv >> 1;        // batch tile 0..3
    const int nt0   = (wv & 1) * 2;   // gate tiles {nt0, nt0+1}

    const unsigned hb_lds = (unsigned)(unsigned long long)(void*)Hb; // LDS byte addr

    // fixed thread<->(b,j) mapping for the elementwise phase; c in registers
    float creg[4] = {0.f, 0.f, 0.f, 0.f};
    int pb[4], pj[4];
#pragma unroll
    for (int k = 0; k < 4; ++k) {
        int p = tid + THREADS * k;    // 0..1023 over (b,j)
        pb[k] = p >> 4;
        pj[k] = p & 15;
    }

    __syncthreads();

    for (int t = 0; t < T_STEPS; ++t) {
        const bf16_t* hc = (t & 1) ? hbuf1 : hbuf0;   // h_t (read)
        bf16_t*       hn = (t & 1) ? hbuf0 : hbuf1;   // h_{t+1} (write)

        // ---- TDM: DMA h_t [64x512 bf16] global -> LDS, 16B pad per row ----
        // pad_interval code 7 = 256 DWORDs (=1024B row), pad_amount code 3 = 4
        // DWORDs (16B) => LDS row stride 1040B = HPITCH*2 (bank-skewed).
        if (tid < 32) {   // wave 0 issues once; EXEC uniform within the wave
            tdm_load_2d_bf16(hb_lds, hc, HID, BATCH, HID, 7u, 3u);
        }

        // ---- stage x_t as bf16 into LDS (overlaps the TDM transfer) ----
        const float4* xs = (const float4*)(x + (size_t)t * BATCH * INF);
        for (int i = tid; i < BATCH * INF / 4; i += THREADS) {
            int b = i >> 6, c4 = i & 63;              // 64 float4 per row
            float4 v = xs[i];
            union { bf16_t b4[4]; unsigned long long u; } pk;
            pk.b4[0] = (bf16_t)v.x; pk.b4[1] = (bf16_t)v.y;
            pk.b4[2] = (bf16_t)v.z; pk.b4[3] = (bf16_t)v.w;
            *(unsigned long long*)(Xb + b * XPITCH + c4 * 4) = pk.u;
        }
        if (t + 1 < T_STEPS) {   // gfx1250 global_prefetch_b8 for next x_t
            const char* nx = (const char*)(x + (size_t)(t + 1) * BATCH * INF);
            __builtin_prefetch(nx + tid * 256, 0, 1);
        }
        if (tid < 32) {
            __builtin_amdgcn_s_wait_tensorcnt(0);
        }
        __syncthreads();

        // ---- WMMA GEMM: gates[64x64] = [x|h][64x768] * W^T slice ----
        v8f acc0 = {}; v8f acc1 = {};
        const bf16_t* arowx = Xb + (mtile * 16 + lm) * XPITCH;
        const bf16_t* arowh = Hb + (mtile * 16 + lm) * HPITCH;
        const bf16_t* brow0 = Wb + ((nt0    ) * 16 + lm) * PITCH;
        const bf16_t* brow1 = Wb + ((nt0 + 1) * 16 + lm) * PITCH;
#pragma unroll 4
        for (int kc = 0; kc < INF; kc += 32) {        // x phase: K = 0..255
            BFrag a, b0, b1;
            a.h[0]  = *(const v8bf*)(arowx + kc + hsel * 16);
            a.h[1]  = *(const v8bf*)(arowx + kc + hsel * 16 + 8);
            b0.h[0] = *(const v8bf*)(brow0 + kc + hsel * 16);
            b0.h[1] = *(const v8bf*)(brow0 + kc + hsel * 16 + 8);
            b1.h[0] = *(const v8bf*)(brow1 + kc + hsel * 16);
            b1.h[1] = *(const v8bf*)(brow1 + kc + hsel * 16 + 8);
            acc0 = __builtin_amdgcn_wmma_f32_16x16x32_bf16(
                       false, a.v, false, b0.v, (short)0, acc0, false, false);
            acc1 = __builtin_amdgcn_wmma_f32_16x16x32_bf16(
                       false, a.v, false, b1.v, (short)0, acc1, false, false);
        }
#pragma unroll 4
        for (int kc = 0; kc < HID; kc += 32) {        // h phase: K = 256..767
            BFrag a, b0, b1;
            a.h[0]  = *(const v8bf*)(arowh + kc + hsel * 16);
            a.h[1]  = *(const v8bf*)(arowh + kc + hsel * 16 + 8);
            b0.h[0] = *(const v8bf*)(brow0 + INF + kc + hsel * 16);
            b0.h[1] = *(const v8bf*)(brow0 + INF + kc + hsel * 16 + 8);
            b1.h[0] = *(const v8bf*)(brow1 + INF + kc + hsel * 16);
            b1.h[1] = *(const v8bf*)(brow1 + INF + kc + hsel * 16 + 8);
            acc0 = __builtin_amdgcn_wmma_f32_16x16x32_bf16(
                       false, a.v, false, b0.v, (short)0, acc0, false, false);
            acc1 = __builtin_amdgcn_wmma_f32_16x16x32_bf16(
                       false, a.v, false, b1.v, (short)0, acc1, false, false);
        }

        // ---- spill D tiles to gates LDS: [b][gate][j] ----
#pragma unroll
        for (int r = 0; r < 8; ++r) {
            int b = mtile * 16 + hsel * 8 + r;
            gatesL[(b * 4 + nt0    ) * 16 + lm] = acc0[r];
            gatesL[(b * 4 + nt0 + 1) * 16 + lm] = acc1[r];
        }
        __syncthreads();

        // ---- elementwise LSTM cell update ----
#pragma unroll
        for (int k = 0; k < 4; ++k) {
            int b = pb[k], j = pj[k], jg = jbase + j;
            float ig = gatesL[(b * 4 + 0) * 16 + j] + bih[jg]           + bhh[jg];
            float fg = gatesL[(b * 4 + 1) * 16 + j] + bih[HID + jg]     + bhh[HID + jg];
            float gg = gatesL[(b * 4 + 2) * 16 + j] + bih[2 * HID + jg] + bhh[2 * HID + jg];
            float og = gatesL[(b * 4 + 3) * 16 + j] + bih[3 * HID + jg] + bhh[3 * HID + jg];
            ig = sigmoidf_fast(ig);
            fg = sigmoidf_fast(fg);
            gg = tanhf(gg);
            og = sigmoidf_fast(og);
            float c = fg * creg[k] + ig * gg;
            creg[k] = c;
            float h = og * tanhf(c);
            hn[b * HID + jg] = (bf16_t)h;
            if (t == T_STEPS - 1) {
                out[OUT_OFF_H + b * HID + jg] = h;   // h_T
                out[OUT_OFF_C + b * HID + jg] = c;   // c_T
            }
        }
        grid_barrier(cnt, phase, ++target);
    }
}

extern "C" void kernel_launch(void* const* d_in, const int* in_sizes, int n_in,
                              void* d_out, int out_size, void* d_ws, size_t ws_size,
                              hipStream_t stream) {
    (void)in_sizes; (void)n_in; (void)out_size; (void)ws_size;
    const float* x   = (const float*)d_in[0];
    const float* Wih = (const float*)d_in[1];
    const float* Whh = (const float*)d_in[2];
    const float* bih = (const float*)d_in[3];
    const float* bhh = (const float*)d_in[4];
    const float* fcw = (const float*)d_in[5];
    const float* fcb = (const float*)d_in[6];

    char* ws = (char*)d_ws;
    bf16_t*  Wcomb = (bf16_t*)ws;                                   // 3 MB
    bf16_t*  h0    = (bf16_t*)(ws + (size_t)4 * HID * KTOT * 2);    // 64 KB
    bf16_t*  h1    = h0 + BATCH * HID;                              // 64 KB
    unsigned* bar  = (unsigned*)((char*)(h1 + BATCH * HID));        // 8 B

    const int prepN = 4 * HID * KTOT;   // 1,572,864 elems
    lstm_prep<<<dim3((prepN + THREADS - 1) / THREADS), dim3(THREADS), 0, stream>>>(
        Wih, Whh, Wcomb, h0, bar);

    lstm_persistent<<<dim3(NWG_TOTAL), dim3(THREADS), LDS_TOTAL, stream>>>(
        x, bih, bhh, fcw, fcb, Wcomb, h0, h1, bar, (float*)d_out);
}
